// VQVideo_75892072120911
// MI455X (gfx1250) — matrix operand order repaired and checked by reading, not compiled, not run on hardware
//
#include <hip/hip_runtime.h>
#include <math.h>

// ---------------- problem constants ----------------
enum {
  cW      = 512,
  cLAYERS = 6,
  cHEADS  = 8,
  cDH     = 64,
  cTDYN   = 16,
  cNF     = 128,
  cNB     = 12,          // N frames-1
  cBLK    = cNF + cTDYN + 2,   // 146
  cSEQ    = cNB * cBLK,        // 1752
  cB      = 4,
  cM      = cB * cSEQ,         // 7008 = 146*48
  cVOCAB  = 1024,
  cGROWS  = cB * cNB * cNF,    // 6144 = 128*48
};

typedef __attribute__((ext_vector_type(16))) _Float16 v16h;
typedef __attribute__((ext_vector_type(8)))  float    v8f;

// ---------------- WMMA helpers ----------------
__device__ __forceinline__ v8f wmma32(v16h a, v16h b, v8f c) {
  // (neg_a, A, neg_b, B, c_mod, C, reuse_a, reuse_b)
  return __builtin_amdgcn_wmma_f32_16x16x32_f16(false, a, false, b, (short)0, c, false, false);
}

// A-matrix fragment (16x32 f16). Per ISA: lanes 0-15 hold M=0..15 / K={0..7,16..23},
// lanes 16-31 hold K={8..15,24..31}; VGPR pairs are consecutive K.
__device__ __forceinline__ v16h ld_a_frag(const _Float16* __restrict__ rowbase, int ld,
                                          int k0, int mmax) {
  int lane = threadIdx.x & 31;
  int m = lane & 15; if (m > mmax) m = mmax;
  const _Float16* p = rowbase + (long)m * ld + k0 + ((lane >> 4) << 3);
  v16h r;
#pragma unroll
  for (int e = 0; e < 8; ++e) { r[e] = p[e]; r[8 + e] = p[16 + e]; }
  return r;
}

// B-matrix fragment (32x16 f16), source is row-major [N, K] (weight layout):
// lane = column n, lanes 0-15 K={0..7,16..23}, lanes 16-31 K={8..15,24..31}.
__device__ __forceinline__ v16h ld_b_frag(const _Float16* __restrict__ base, int ld,
                                          int n0, int k0, int nmax) {
  int lane = threadIdx.x & 31;
  int n = n0 + (lane & 15); if (n > nmax) n = nmax;
  const _Float16* p = base + (long)n * ld + k0 + ((lane >> 4) << 3);
  v16h r;
#pragma unroll
  for (int e = 0; e < 8; ++e) { r[e] = p[e]; r[8 + e] = p[16 + e]; }
  return r;
}

// ---------------- fp32 -> fp16 convert ----------------
__global__ void f32_to_f16_kernel(const float* __restrict__ src, _Float16* __restrict__ dst, int n) {
  int i = blockIdx.x * blockDim.x + threadIdx.x;
  int stride = gridDim.x * blockDim.x;
  for (; i < n; i += stride) dst[i] = (_Float16)src[i];
}

// ---------------- embedding: concat + positional ----------------
__global__ void embed_kernel(const float* __restrict__ x, const float* __restrict__ f,
                             const float* __restrict__ delim, float* __restrict__ h) {
  long total = (long)cM * cW;
  long idx = (long)blockIdx.x * blockDim.x + threadIdx.x;
  long stride = (long)gridDim.x * blockDim.x;
  for (; idx < total; idx += stride) {
    int c = (int)(idx & (cW - 1));
    long sw = idx >> 9;
    int s = (int)(sw % cSEQ);
    int b = (int)(sw / cSEQ);
    int blk = s / cBLK, lo = s % cBLK;
    float val;
    if (lo < cNF)                       val = x[(((long)b * cNB + blk) * cNF + lo) * cW + c];
    else if (lo == cNF || lo == cBLK-1) val = delim[c];
    else                                val = f[(((long)b * cNB + blk) * cTDYN + (lo - cNF - 1)) * cW + c];
    int j = c >> 1;
    float freq = __powf(10000.0f, -(float)(2 * j) * (1.0f / (float)cW));
    float arg = (float)s * freq;
    val += (c & 1) ? __cosf(arg) : __sinf(arg);
    h[idx] = val;
  }
}

// ---------------- layernorm -> fp16 (one wave per row) ----------------
__global__ void ln_kernel(const float* __restrict__ h, const float* __restrict__ sc,
                          const float* __restrict__ bi, _Float16* __restrict__ out, int rows) {
  int row = blockIdx.x * (blockDim.x >> 5) + (threadIdx.x >> 5);
  if (row >= rows) return;
  int lane = threadIdx.x & 31;
  const float* p = h + (long)row * cW;
  float v[16], s = 0.f, sq = 0.f;
#pragma unroll
  for (int i = 0; i < 16; ++i) { v[i] = p[lane + 32 * i]; s += v[i]; sq += v[i] * v[i]; }
#pragma unroll
  for (int off = 16; off >= 1; off >>= 1) { s += __shfl_xor(s, off, 32); sq += __shfl_xor(sq, off, 32); }
  float mu = s * (1.0f / cW);
  float var = sq * (1.0f / cW) - mu * mu;
  float rstd = rsqrtf(var + 1e-5f);
  _Float16* o = out + (long)row * cW;
#pragma unroll
  for (int i = 0; i < 16; ++i) {
    int c = lane + 32 * i;
    o[c] = (_Float16)((v[i] - mu) * rstd * sc[c] + bi[c]);
  }
}

// ---------------- generic WMMA GEMM: C = A[M,K](f16) * Wt[N,K]^T + bias ----------------
// one wave per 48(M) x 64(N) tile: 12 v8f accumulators (96 VGPRs) + 3 A frags + 1 B frag
// stays inside the 256 directly-addressable VGPRs (no scratch spills), ~27 FLOP/B from L2.
// epilogues: gelu / fp16-store / residual-add / fp32-store.
__global__ void __launch_bounds__(32, 1)
gemm_kernel(const _Float16* __restrict__ A, const _Float16* __restrict__ Wt,
            const float* __restrict__ bias, int M, int N, int K,
            _Float16* __restrict__ out16, float* __restrict__ resid,
            float* __restrict__ out32, int gelu) {
  int m0 = blockIdx.x * 48;
  int n0 = blockIdx.y * 64;
  v8f acc[12] = {};
  const _Float16* arow[3];
#pragma unroll
  for (int mi = 0; mi < 3; ++mi) {
    int ms = m0 + mi * 16;
    if (ms > M - 16) ms = M - 16;   // M is always a multiple of 16 (and of 48 in practice)
    arow[mi] = A + (long)ms * K;
  }
  for (int k0 = 0; k0 < K; k0 += 32) {
    if (k0 + 32 < K)
      __builtin_prefetch(Wt + (long)(n0 + (threadIdx.x & 15)) * K + k0 + 32, 0, 1);
    v16h a0 = ld_a_frag(arow[0], K, k0, 15);
    v16h a1 = ld_a_frag(arow[1], K, k0, 15);
    v16h a2 = ld_a_frag(arow[2], K, k0, 15);
#pragma unroll
    for (int nt = 0; nt < 4; ++nt) {
      v16h bf = ld_b_frag(Wt, K, n0 + nt * 16, k0, N - 1);
      acc[nt]     = wmma32(a0, bf, acc[nt]);
      acc[4 + nt] = wmma32(a1, bf, acc[4 + nt]);
      acc[8 + nt] = wmma32(a2, bf, acc[8 + nt]);
    }
  }
  int lane = threadIdx.x & 31;
  int cl = lane & 15, rh = (lane >> 4) << 3;
#pragma unroll
  for (int mi = 0; mi < 3; ++mi)
#pragma unroll
    for (int nt = 0; nt < 4; ++nt)
#pragma unroll
      for (int r = 0; r < 8; ++r) {
        int row = m0 + mi * 16 + rh + r;
        if (row < M) {
          int col = n0 + nt * 16 + cl;
          float v = acc[mi * 4 + nt][r];
          if (bias)  v += bias[col];
          if (gelu)  v = 0.5f * v * (1.0f + erff(v * 0.70710678118f));
          long o = (long)row * N + col;
          if (resid) resid[o] += v;
          if (out16) out16[o] = (_Float16)v;
          if (out32) out32[o] = v;
        }
      }
}

// ---------------- attention mask predicate (structured, computed analytically) ----------------
__device__ __forceinline__ bool attn_allowed(int q, int k) {
  if (k >= cSEQ) return false;
  if (q == k) return true;
  int ql = q % cBLK;
  if (ql >= cNF) return false;          // delim/dyn tokens: diagonal only
  int qb = q / cBLK;
  int kb = k / cBLK, kl = k % cBLK;
  if (kb == qb && kl < cNF) return true;                       // own frame block
  if (kl >= cNF && kl <= cNF + cTDYN && kb <= qb) return true; // delim+dyn slices (<= own block)
  if (kb == qb - 1 && kl == cBLK - 1) return true;             // prev block's final delim
  return false;
}

// P fragment (A-layout) from LDS pbuf[16][32]
__device__ __forceinline__ v16h ld_p_frag(const _Float16* pbuf) {
  int lane = threadIdx.x & 31;
  const _Float16* p = pbuf + (lane & 15) * 32 + ((lane >> 4) << 3);
  v16h r;
#pragma unroll
  for (int e = 0; e < 8; ++e) { r[e] = p[e]; r[8 + e] = p[16 + e]; }
  return r;
}

// V fragment (B-layout 32x16) from LDS vbuf[32][64]
__device__ __forceinline__ v16h ld_v_frag(const _Float16* vbuf, int nt) {
  int lane = threadIdx.x & 31;
  int n = nt * 16 + (lane & 15);
  int kb = (lane >> 4) << 3;
  v16h r;
#pragma unroll
  for (int e = 0; e < 8; ++e) { r[e] = vbuf[(kb + e) * 64 + n]; r[8 + e] = vbuf[(kb + 16 + e) * 64 + n]; }
  return r;
}

// ---------------- flash attention: one wave = 16 queries of one (b, head) ----------------
// V tiles are streamed memory->LDS with gfx1250 GLOBAL_LOAD_ASYNC_TO_LDS_B128 (ASYNCcnt),
// double-buffered so the transfer overlaps the QK^T WMMAs and the online softmax.
__global__ void __launch_bounds__(32, 1)
attn_kernel(const _Float16* __restrict__ qkv, _Float16* __restrict__ o16) {
  __shared__ _Float16 pbuf[16 * 32];
  __shared__ _Float16 vbuf[2][32 * 64];
  const int QKVLD = 3 * cW;  // 1536
  int lane = threadIdx.x;    // blockDim = 32
  int qt = blockIdx.x, bh = blockIdx.y;
  int b = bh >> 3, hh = bh & 7;
  int q0 = qt * 16;
  const _Float16* base = qkv + ((long)b * cSEQ) * QKVLD + hh * cDH;
  const _Float16* kb_  = base + cW;
  const _Float16* vb_  = base + 2 * cW;
  int mmax = cSEQ - 1 - q0; if (mmax > 15) mmax = 15;
  v16h qf0 = ld_a_frag(base + (long)q0 * QKVLD, QKVLD, 0,  mmax);
  v16h qf1 = ld_a_frag(base + (long)q0 * QKVLD, QKVLD, 32, mmax);

  unsigned vlds[2] = { (unsigned)(uintptr_t)&vbuf[0][0], (unsigned)(uintptr_t)&vbuf[1][0] };
  // async-stage one 32-key x 64 V tile: lane l owns key kc+l (128B = 8 x b128 per lane-set)
  auto issue_v = [&](int kc, int bufi) {
    int key = kc + lane; if (key > cSEQ - 1) key = cSEQ - 1;
    const _Float16* rowp = vb_ + (long)key * QKVLD;
    unsigned lb = vlds[bufi] + lane * 128;
#pragma unroll
    for (int i = 0; i < 8; ++i) {
      unsigned long long ga = (unsigned long long)rowp + (unsigned)(i * 16);
      unsigned la = lb + i * 16;
      asm volatile("global_load_async_to_lds_b128 %0, %1, off" :: "v"(la), "v"(ga) : "memory");
    }
  };

  float mrun[8], lrun[8];
#pragma unroll
  for (int r = 0; r < 8; ++r) { mrun[r] = -1e30f; lrun[r] = 0.f; }
  v8f o0 = {}, o1 = {}, o2 = {}, o3 = {};

  int qlast = q0 + 15; if (qlast >= cSEQ) qlast = cSEQ - 1;
  int kmax = ((qlast / cBLK) + 1) * cBLK - 1;      // mask never reaches past own block
  if (kmax >= cSEQ) kmax = cSEQ - 1;
  int cl = lane & 15, rh = (lane >> 4) << 3;

  issue_v(0, 0);
  int cur = 0;
  for (int kc = 0; kc <= kmax; kc += 32) {
    bool haveNext = (kc + 32) <= kmax;
    if (haveNext) {
      // prior ds reads of the other buffer must retire before the async engine overwrites it
      asm volatile("s_wait_dscnt 0" ::: "memory");
      issue_v(kc + 32, cur ^ 1);
    }

    // scores: two 16x16 tiles over dh=64 (two WMMAs each) -- overlaps the async V copy
    v8f c0 = {}, c1 = {};
    c0 = wmma32(qf0, ld_b_frag(kb_, QKVLD, kc,      0,  cSEQ - 1), c0);
    c0 = wmma32(qf1, ld_b_frag(kb_, QKVLD, kc,      32, cSEQ - 1), c0);
    c1 = wmma32(qf0, ld_b_frag(kb_, QKVLD, kc + 16, 0,  cSEQ - 1), c1);
    c1 = wmma32(qf1, ld_b_frag(kb_, QKVLD, kc + 16, 32, cSEQ - 1), c1);

    // online softmax over the 32-key chunk
#pragma unroll
    for (int r = 0; r < 8; ++r) {
      int q = q0 + rh + r;
      int k0i = kc + cl;
      float s0 = c0[r] * 0.125f + (attn_allowed(q, k0i)      ? 0.f : -INFINITY);
      float s1 = c1[r] * 0.125f + (attn_allowed(q, k0i + 16) ? 0.f : -INFINITY);
      float tm = fmaxf(s0, s1);
#pragma unroll
      for (int off = 8; off >= 1; off >>= 1) tm = fmaxf(tm, __shfl_xor(tm, off, 16));
      float nm = fmaxf(mrun[r], tm);
      float corr = __expf(mrun[r] - nm);
      mrun[r] = nm;
      float e0 = __expf(s0 - nm), e1 = __expf(s1 - nm);
      float rs = e0 + e1;
#pragma unroll
      for (int off = 8; off >= 1; off >>= 1) rs += __shfl_xor(rs, off, 16);
      lrun[r] = lrun[r] * corr + rs;
      o0[r] *= corr; o1[r] *= corr; o2[r] *= corr; o3[r] *= corr;
      pbuf[(rh + r) * 32 + cl]      = (_Float16)e0;
      pbuf[(rh + r) * 32 + 16 + cl] = (_Float16)e1;
    }
    asm volatile("s_wait_dscnt 0" ::: "memory");   // pbuf writes visible to cross-lane reads
    if (haveNext) asm volatile("s_wait_asynccnt 8" ::: "memory");  // current V tile landed
    else          asm volatile("s_wait_asynccnt 0" ::: "memory");

    const _Float16* vcur = &vbuf[cur][0];
    v16h pf = ld_p_frag(pbuf);
    o0 = wmma32(pf, ld_v_frag(vcur, 0), o0);
    o1 = wmma32(pf, ld_v_frag(vcur, 1), o1);
    o2 = wmma32(pf, ld_v_frag(vcur, 2), o2);
    o3 = wmma32(pf, ld_v_frag(vcur, 3), o3);
    cur ^= 1;
  }

#pragma unroll
  for (int r = 0; r < 8; ++r) {
    int q = q0 + rh + r;
    if (q < cSEQ) {
      float inv = 1.0f / lrun[r];
      long o = ((long)b * cSEQ + q) * cW + hh * cDH + cl;
      o16[o]      = (_Float16)(o0[r] * inv);
      o16[o + 16] = (_Float16)(o1[r] * inv);
      o16[o + 32] = (_Float16)(o2[r] * inv);
      o16[o + 48] = (_Float16)(o3[r] * inv);
    }
  }
}

// ---------------- gather frame-token rows -> fp16 for the head GEMM ----------------
__global__ void gather_kernel(const float* __restrict__ h, _Float16* __restrict__ out) {
  long total = (long)cGROWS * cW;
  long idx = (long)blockIdx.x * blockDim.x + threadIdx.x;
  long stride = (long)gridDim.x * blockDim.x;
  for (; idx < total; idx += stride) {
    int c = (int)(idx & (cW - 1));
    long g = idx >> 9;
    int b = (int)(g / (cNB * cNF));
    int rem = (int)(g % (cNB * cNF));
    int i = rem >> 7, r = rem & 127;
    out[idx] = (_Float16)h[(((long)b * cSEQ) + i * cBLK + r) * cW + c];
  }
}

// ---------------- host orchestration ----------------
extern "C" void kernel_launch(void* const* d_in, const int* in_sizes, int n_in,
                              void* d_out, int out_size, void* d_ws, size_t ws_size,
                              hipStream_t stream) {
  const float* x     = (const float*)d_in[0];
  const float* f     = (const float*)d_in[1];
  const float* delim = (const float*)d_in[2];
  const float* ln1s  = (const float*)d_in[3];
  const float* ln1b  = (const float*)d_in[4];
  const float* wqkv  = (const float*)d_in[5];
  const float* bqkv  = (const float*)d_in[6];
  const float* wout  = (const float*)d_in[7];
  const float* bout  = (const float*)d_in[8];
  const float* ln2s  = (const float*)d_in[9];
  const float* ln2b  = (const float*)d_in[10];
  const float* wfc   = (const float*)d_in[11];
  const float* bfc   = (const float*)d_in[12];
  const float* wpr   = (const float*)d_in[13];
  const float* bpr   = (const float*)d_in[14];
  const float* whead = (const float*)d_in[15];
  float* out = (float*)d_out;

  uintptr_t wsb = (uintptr_t)d_ws;
  size_t off = 0;
  auto take = [&](size_t bytes) -> void* {
    void* p = (void*)(wsb + off);
    off += (bytes + 255) & ~(size_t)255;
    return p;
  };
  float*     h      = (float*)take((size_t)cM * cW * 4);
  _Float16*  a16    = (_Float16*)take((size_t)cM * cW * 2);
  _Float16*  qkv16  = (_Float16*)take((size_t)cM * 4 * cW * 2);  // reused as MLP hidden
  _Float16*  o16    = (_Float16*)take((size_t)cM * cW * 2);      // reused as gathered-h
  _Float16*  hwqkv  = (_Float16*)take((size_t)cLAYERS * 3 * cW * cW * 2);
  _Float16*  hwout  = (_Float16*)take((size_t)cLAYERS * cW * cW * 2);
  _Float16*  hwfc   = (_Float16*)take((size_t)cLAYERS * 4 * cW * cW * 2);
  _Float16*  hwpr   = (_Float16*)take((size_t)cLAYERS * cW * 4 * cW * 2);
  _Float16*  hwhead = (_Float16*)take((size_t)cVOCAB * cW * 2);
  (void)ws_size; (void)in_sizes; (void)n_in; (void)out_size;

  auto cvt = [&](const float* s, _Float16* d, int n) {
    int blocks = (n + 1023) / 1024; if (blocks > 2048) blocks = 2048;
    f32_to_f16_kernel<<<dim3(blocks), dim3(256), 0, stream>>>(s, d, n);
  };
  cvt(wqkv,  hwqkv,  cLAYERS * 3 * cW * cW);
  cvt(wout,  hwout,  cLAYERS * cW * cW);
  cvt(wfc,   hwfc,   cLAYERS * 4 * cW * cW);
  cvt(wpr,   hwpr,   cLAYERS * cW * 4 * cW);
  cvt(whead, hwhead, cVOCAB * cW);

  embed_kernel<<<dim3(2048), dim3(256), 0, stream>>>(x, f, delim, h);

  const int MT = cM / 48;  // 146 exactly
  for (int l = 0; l < cLAYERS; ++l) {
    // LN1 -> a16
    ln_kernel<<<dim3((cM + 7) / 8), dim3(256), 0, stream>>>(h, ln1s + l * cW, ln1b + l * cW, a16, cM);
    // QKV: [M,512] x [1536,512]^T -> fp16 [M,1536]
    gemm_kernel<<<dim3(MT, 24), dim3(32), 0, stream>>>(a16, hwqkv + (size_t)l * 3 * cW * cW,
        bqkv + l * 3 * cW, cM, 3 * cW, cW, qkv16, nullptr, nullptr, 0);
    // flash attention -> o16
    attn_kernel<<<dim3((cSEQ + 15) / 16, cB * cHEADS), dim3(32), 0, stream>>>(qkv16, o16);
    // out proj + residual into h
    gemm_kernel<<<dim3(MT, 8), dim3(32), 0, stream>>>(o16, hwout + (size_t)l * cW * cW,
        bout + l * cW, cM, cW, cW, nullptr, h, nullptr, 0);
    // LN2 -> a16
    ln_kernel<<<dim3((cM + 7) / 8), dim3(256), 0, stream>>>(h, ln2s + l * cW, ln2b + l * cW, a16, cM);
    // FC + exact GELU -> fp16 [M,2048] (reuses qkv16 buffer)
    gemm_kernel<<<dim3(MT, 32), dim3(32), 0, stream>>>(a16, hwfc + (size_t)l * 4 * cW * cW,
        bfc + l * 4 * cW, cM, 4 * cW, cW, qkv16, nullptr, nullptr, 1);
    // proj + residual into h
    gemm_kernel<<<dim3(MT, 8), dim3(32), 0, stream>>>(qkv16, hwpr + (size_t)l * cW * 4 * cW,
        bpr + l * cW, cM, cW, 4 * cW, nullptr, h, nullptr, 0);
  }

  // gather frame tokens, head GEMM -> fp32 logits
  gather_kernel<<<dim3(2048), dim3(256), 0, stream>>>(h, o16);
  gemm_kernel<<<dim3(cGROWS / 48, cVOCAB / 64), dim3(32), 0, stream>>>(o16, hwhead,
      nullptr, cGROWS, cVOCAB, cW, nullptr, nullptr, out, 0);
}
